// LinearDepthWiseAttention_28432683499777
// MI455X (gfx1250) — matrix operand order repaired and checked by reading, not compiled, not run on hardware
//
#include <hip/hip_runtime.h>

// ---------------------------------------------------------------------------
// LinearDepthWiseAttention for MI455X (gfx1250, wave32, WMMA)
//
//   K0a: w_qkv f32 -> bf16                        (wb)
//   K0b: net (b,c,n) f32 -> netb (b,n,c) bf16     [LDS tiled transpose]
//   K1 : QKV GEMM, 32x64 tile/wave (v_wmma_f32_16x16x32_bf16)
//          q -> qT   (b, n, 512)  l-contiguous
//          k -> kbuf (b, 512, n)  n-contiguous
//          v -> vbuf (b, 512, n)  n-contiguous
//   K2 : softmax over kbuf rows (f32 math, bf16 storage)
//   K3 : context GEMM, 16x64 tile/wave, split-K(n) x4 -> f32 partials
//   K4 : fold  Wc[b,c,(h,d)] = sum_e w_out[c,(h,e)] * sum_p ctxp[...]
//   K5 : final GEMM, 32x64 tile/wave: out = Wc . q + b_out
// ---------------------------------------------------------------------------

typedef __bf16 bf16_t;
typedef __attribute__((ext_vector_type(16))) __bf16 v16bf;
typedef __attribute__((ext_vector_type(8)))  __bf16 v8bf;
typedef __attribute__((ext_vector_type(8)))  float  v8f;

#define BATCH   16
#define CIN     256
#define NPIX    4096      // 64*64
#define LAT     512       // 8 heads * 64
#define OQKV    1536      // 3*LAT
#define CTXP    ((size_t)BATCH * 8 * 64 * 64)   // elems per context partial

// ---- WMMA fragment loaders (bf16, 16x16x32) -------------------------------
// A (16x32, MxK): lane m = lane&15, half = lane>>4.
//   elem e -> k = half*16 + e
__device__ __forceinline__ v16bf load_a_frag(const bf16_t* __restrict__ row, int half) {
  return *(const v16bf*)(row + half * 16);
}
// B (32x16, KxN): lane col = lane&15, half = lane>>4; elem e -> k = half*16+e.
__device__ __forceinline__ v16bf load_b_frag(const bf16_t* __restrict__ col, int half) {
  return *(const v16bf*)(col + half * 16);
}
#define WMMA_BF16(A, Bf, Cc) \
  __builtin_amdgcn_wmma_f32_16x16x32_bf16(false, (A), false, (Bf), (short)0, (Cc), false, false)

// ---- K0a: elementwise f32 -> bf16 -----------------------------------------
__global__ __launch_bounds__(256) void convert_f32_bf16(const float* __restrict__ src,
                                                        bf16_t* __restrict__ dst, int n) {
  int i = blockIdx.x * 256 + threadIdx.x;
  if (i < n) dst[i] = (bf16_t)src[i];
}

// ---- K0b: (b,c,n) f32 -> (b,n,c) bf16, 32x32 LDS tiled --------------------
__global__ __launch_bounds__(256) void transpose_net(const float* __restrict__ net,
                                                     bf16_t* __restrict__ netb) {
  __shared__ float tile[32][33];
  const int b  = blockIdx.z;
  const int c0 = blockIdx.y * 32;
  const int n0 = blockIdx.x * 32;
  const int tx = threadIdx.x & 31;
  const int ty = threadIdx.x >> 5;   // 0..7
#pragma unroll
  for (int i = ty; i < 32; i += 8)
    tile[i][tx] = net[((size_t)b * CIN + c0 + i) * NPIX + n0 + tx];
  __syncthreads();
#pragma unroll
  for (int i = ty; i < 32; i += 8)
    netb[((size_t)b * NPIX + n0 + i) * CIN + c0 + tx] = (bf16_t)tile[tx][i];
}

// ---- K1: QKV GEMM. one wave per 32(o) x 64(n) tile, K=256 -----------------
__global__ __launch_bounds__(256) void qkv_gemm(const bf16_t* __restrict__ wb,    // (1536,256)
                                                const bf16_t* __restrict__ netb,  // (B,N,C)
                                                bf16_t* __restrict__ qT,          // (B,N,512)
                                                bf16_t* __restrict__ kbuf,        // (B,512,N)
                                                bf16_t* __restrict__ vbuf) {      // (B,512,N)
  const int lane = threadIdx.x & 31;
  const int wave = threadIdx.x >> 5;
  int tile = blockIdx.x * 8 + wave;          // 16 * 48 * 64 waves
  const int nT = tile & 63; tile >>= 6;
  const int oT = tile % 48;
  const int b  = tile / 48;
  const int half = lane >> 4, l16 = lane & 15;
  const int o0 = oT * 32, n0 = nT * 64;

  const bf16_t* arow = wb   + (size_t)(o0 + l16) * CIN;
  const bf16_t* brow = netb + ((size_t)b * NPIX + n0 + l16) * CIN;

  v8f acc[2][4] = {};
#pragma unroll
  for (int kk = 0; kk < CIN / 32; ++kk) {
    const v16bf a0 = load_a_frag(arow + kk * 32, half);
    const v16bf a1 = load_a_frag(arow + (size_t)16 * CIN + kk * 32, half);
#pragma unroll
    for (int ni = 0; ni < 4; ++ni) {
      const v16bf bb = load_b_frag(brow + (size_t)ni * 16 * CIN + kk * 32, half);
      acc[0][ni] = WMMA_BF16(a0, bb, acc[0][ni]);
      acc[1][ni] = WMMA_BF16(a1, bb, acc[1][ni]);
    }
  }
  // D layout: elem r -> (m = o0 + oi*16 + half*8 + r, n = n0 + ni*16 + l16)
  if (o0 < LAT) {                 // q: transposed store, 8 contiguous bf16
#pragma unroll
    for (int oi = 0; oi < 2; ++oi)
#pragma unroll
      for (int ni = 0; ni < 4; ++ni) {
        v8bf pk;
#pragma unroll
        for (int r = 0; r < 8; ++r) pk[r] = (bf16_t)acc[oi][ni][r];
        *(v8bf*)(qT + ((size_t)b * NPIX + n0 + ni * 16 + l16) * LAT
                    + o0 + oi * 16 + half * 8) = pk;
      }
  } else {
    bf16_t* base = (o0 < 2 * LAT) ? kbuf : vbuf;
    const int lb = o0 - ((o0 < 2 * LAT) ? LAT : 2 * LAT);
#pragma unroll
    for (int oi = 0; oi < 2; ++oi)
#pragma unroll
      for (int ni = 0; ni < 4; ++ni)
#pragma unroll
        for (int r = 0; r < 8; ++r) {
          const int l = lb + oi * 16 + half * 8 + r;
          base[((size_t)b * LAT + l) * NPIX + n0 + ni * 16 + l16] =
              (bf16_t)acc[oi][ni][r];
        }
  }
}

// ---- K2: row softmax over kbuf (one block per row) ------------------------
__global__ __launch_bounds__(256) void softmax_rows(bf16_t* __restrict__ kbuf) {
  bf16_t* p = kbuf + (size_t)blockIdx.x * NPIX;
  __shared__ float red[256];
  float m = -3.4e38f;
  for (int i = threadIdx.x; i < NPIX; i += 256) m = fmaxf(m, (float)p[i]);
  red[threadIdx.x] = m; __syncthreads();
  for (int s = 128; s > 0; s >>= 1) {
    if (threadIdx.x < s) red[threadIdx.x] = fmaxf(red[threadIdx.x], red[threadIdx.x + s]);
    __syncthreads();
  }
  m = red[0]; __syncthreads();
  float sum = 0.f;
  for (int i = threadIdx.x; i < NPIX; i += 256) sum += __expf((float)p[i] - m);
  red[threadIdx.x] = sum; __syncthreads();
  for (int s = 128; s > 0; s >>= 1) {
    if (threadIdx.x < s) red[threadIdx.x] += red[threadIdx.x + s];
    __syncthreads();
  }
  const float inv = 1.0f / red[0];
  for (int i = threadIdx.x; i < NPIX; i += 256)
    p[i] = (bf16_t)(__expf((float)p[i] - m) * inv);
}

// ---- K3: context GEMM, split-K. ctxp[p,b,h,d,e] partial over n-chunk p ----
__global__ __launch_bounds__(256) void context_gemm(const bf16_t* __restrict__ kbuf,
                                                    const bf16_t* __restrict__ vbuf,
                                                    float* __restrict__ ctxp) { // (4,B,8,64,64)
  const int lane = threadIdx.x & 31;
  const int wave = threadIdx.x >> 5;
  int tile = blockIdx.x * 8 + wave;          // 16*8*4*4 = 2048 waves
  const int part = tile & 3; tile >>= 2;
  const int dT   = tile & 3; tile >>= 2;
  const int h    = tile & 7;
  const int b    = tile >> 3;
  const int half = lane >> 4, l16 = lane & 15;
  const int nbase = part * (NPIX / 4);       // 1024-wide K chunk

  const bf16_t* arow = kbuf + ((size_t)b * LAT + h * 64 + dT * 16 + l16) * NPIX + nbase;
  const bf16_t* brow = vbuf + ((size_t)b * LAT + h * 64 + l16) * NPIX + nbase;

  v8f acc[4] = {};
#pragma unroll 2
  for (int kk = 0; kk < (NPIX / 4) / 32; ++kk) {
    const v16bf a = load_a_frag(arow + kk * 32, half);
#pragma unroll
    for (int ni = 0; ni < 4; ++ni) {
      const v16bf bb = load_b_frag(brow + (size_t)ni * 16 * NPIX + kk * 32, half);
      acc[ni] = WMMA_BF16(a, bb, acc[ni]);
    }
  }
  float* out = ctxp + (size_t)part * CTXP + (((size_t)b * 8 + h) * 64) * 64;
#pragma unroll
  for (int ni = 0; ni < 4; ++ni)
#pragma unroll
    for (int r = 0; r < 8; ++r)
      out[(dT * 16 + half * 8 + r) * 64 + ni * 16 + l16] = acc[ni][r];
}

// ---- K4: fold partial contexts into w_out ---------------------------------
__global__ __launch_bounds__(256) void fold_wout(const float* __restrict__ w_out, // (256,512)
                                                 const float* __restrict__ ctxp,  // (4,B,8,64,64)
                                                 bf16_t* __restrict__ Wc) {       // (B,256,512)
  int idx = blockIdx.x * 256 + threadIdx.x;   // B*256*512
  const int l = idx & 511; idx >>= 9;
  const int c = idx & 255;
  const int b = idx >> 8;
  const int h = l >> 6, d = l & 63;
  const float* wrow = w_out + (size_t)c * LAT + h * 64;
  const float* crow = ctxp + (((size_t)b * 8 + h) * 64 + d) * 64;
  float s = 0.f;
#pragma unroll 8
  for (int e = 0; e < 64; ++e) {
    const float cv = crow[e] + crow[CTXP + e] + crow[2 * CTXP + e] + crow[3 * CTXP + e];
    s += wrow[e] * cv;
  }
  Wc[((size_t)b * 256 + c) * LAT + l] = (bf16_t)s;
}

// ---- K5: final GEMM. 32(c) x 64(n) tile/wave, K=512 -----------------------
__global__ __launch_bounds__(256) void out_gemm(const bf16_t* __restrict__ Wc,   // (B,256,512)
                                                const bf16_t* __restrict__ qT,   // (B,N,512)
                                                const float* __restrict__ b_out, // (256)
                                                float* __restrict__ out) {       // (B,256,N)
  const int lane = threadIdx.x & 31;
  const int wave = threadIdx.x >> 5;
  int tile = blockIdx.x * 8 + wave;          // 16*8*64 waves
  const int nT = tile & 63; tile >>= 6;
  const int cT = tile & 7;
  const int b  = tile >> 3;
  const int half = lane >> 4, l16 = lane & 15;
  const int c0 = cT * 32, n0 = nT * 64;

  const bf16_t* arow = Wc + ((size_t)b * 256  + c0 + l16) * LAT;
  const bf16_t* brow = qT + ((size_t)b * NPIX + n0 + l16) * LAT;

  v8f acc[2][4] = {};
#pragma unroll 4
  for (int kk = 0; kk < LAT / 32; ++kk) {
    const v16bf a0 = load_a_frag(arow + kk * 32, half);
    const v16bf a1 = load_a_frag(arow + (size_t)16 * LAT + kk * 32, half);
#pragma unroll
    for (int ni = 0; ni < 4; ++ni) {
      const v16bf bb = load_b_frag(brow + (size_t)ni * 16 * LAT + kk * 32, half);
      acc[0][ni] = WMMA_BF16(a0, bb, acc[0][ni]);
      acc[1][ni] = WMMA_BF16(a1, bb, acc[1][ni]);
    }
  }
#pragma unroll
  for (int oi = 0; oi < 2; ++oi)
#pragma unroll
    for (int ni = 0; ni < 4; ++ni)
#pragma unroll
      for (int r = 0; r < 8; ++r) {
        const int c = c0 + oi * 16 + half * 8 + r;
        out[((size_t)b * 256 + c) * NPIX + n0 + ni * 16 + l16] =
            acc[oi][ni][r] + b_out[c];
      }
}

// ---------------------------------------------------------------------------
extern "C" void kernel_launch(void* const* d_in, const int* in_sizes, int n_in,
                              void* d_out, int out_size, void* d_ws, size_t ws_size,
                              hipStream_t stream) {
  (void)in_sizes; (void)n_in; (void)out_size; (void)ws_size;
  const float* net   = (const float*)d_in[0];  // (16,256,64,64)
  const float* w_qkv = (const float*)d_in[1];  // (1536,256)
  const float* w_out = (const float*)d_in[2];  // (256,512)
  const float* b_out = (const float*)d_in[3];  // (256)
  float* out = (float*)d_out;                  // (16,256,64,64)

  char* ws = (char*)d_ws;
  size_t off = 0;
  bf16_t* qT   = (bf16_t*)(ws + off); off += (size_t)BATCH * NPIX * LAT * 2;  // 64 MiB
  bf16_t* kbuf = (bf16_t*)(ws + off); off += (size_t)BATCH * LAT * NPIX * 2;  // 64 MiB
  bf16_t* vbuf = (bf16_t*)(ws + off); off += (size_t)BATCH * LAT * NPIX * 2;  // 64 MiB
  bf16_t* netb = (bf16_t*)(ws + off); off += (size_t)BATCH * NPIX * CIN * 2;  // 32 MiB
  bf16_t* wb   = (bf16_t*)(ws + off); off += (size_t)OQKV * CIN * 2;          // 768 KiB
  float*  ctxp = (float*) (ws + off); off += 4 * CTXP * 4;                    // 8 MiB
  bf16_t* Wc   = (bf16_t*)(ws + off); off += (size_t)BATCH * 256 * LAT * 2;   // 4 MiB

  // K0: conversions / transpose
  convert_f32_bf16<<<(OQKV * CIN + 255) / 256, 256, 0, stream>>>(w_qkv, wb, OQKV * CIN);
  transpose_net<<<dim3(NPIX / 32, CIN / 32, BATCH), 256, 0, stream>>>(net, netb);

  // K1: QKV projection. 16*48*64 waves, 8 waves/block
  qkv_gemm<<<(BATCH * 48 * 64) / 8, 256, 0, stream>>>(wb, netb, qT, kbuf, vbuf);

  // K2: softmax over k rows
  softmax_rows<<<BATCH * LAT, 256, 0, stream>>>(kbuf);

  // K3: context = k . v^T, split-K x4 (2048 waves)
  context_gemm<<<(BATCH * 8 * 4 * 4) / 8, 256, 0, stream>>>(kbuf, vbuf, ctxp);

  // K4: fold partial contexts into output weights
  fold_wout<<<(BATCH * 256 * LAT) / 256, 256, 0, stream>>>(w_out, ctxp, Wc);

  // K5: final projection (16*8*64 waves)
  out_gemm<<<(BATCH * 8 * 64) / 8, 256, 0, stream>>>(Wc, qT, b_out, out);
}
